// SD_34437047780053
// MI455X (gfx1250) — matrix operand order repaired and checked by reading, not compiled, not run on hardware
//
#include <hip/hip_runtime.h>
#include <hip/hip_bf16.h>

#define NN    1500
#define NE    4096
#define NG    96
#define DN    64
#define DEI   16
#define DHID  128
#define NB    2
#define TST   3
#define HEADS 2
#define DHEAD 128

typedef _Float16 h16;
typedef __attribute__((ext_vector_type(16))) _Float16 v16h;
typedef __attribute__((ext_vector_type(8)))  float    v8f;
typedef __attribute__((ext_vector_type(4)))  float    f4;

__device__ __forceinline__ float reluf(float x){ return x > 0.f ? x : 0.f; }
__device__ __forceinline__ float eluf (float x){ return x > 0.f ? x : (__expf(x) - 1.f); }

// ---------------------------------------------------------------------------
// f32 [KxN] -> f16 transposed [NxK] weight conversion (done once per launch)
// ---------------------------------------------------------------------------
__global__ void k_f32h_t(const float* __restrict__ src, h16* __restrict__ dst, int K, int N){
    int i = blockIdx.x * blockDim.x + threadIdx.x;
    if (i >= K * N) return;
    int k = i / N, n = i - k * N;
    dst[n * K + k] = (h16)src[i];
}

// ---------------------------------------------------------------------------
// One-wave WMMA GEMM: C[MxN] = act(A[MxK] @ W[KxN] + bias + add)
// W is pre-transposed f16: Wt[N x K]. block = 32 threads (1 wave32),
// each block owns one 16x16 C tile. LDS-free: A via 4x b128 loads per K-tile,
// B via contiguous 32B load of the transposed weights.
// ---------------------------------------------------------------------------
template<int ACT>  // 0=none 1=relu 2=elu
__global__ void __launch_bounds__(32)
k_wmma_gemm(const float* __restrict__ A, const h16* __restrict__ Wt,
            const float* __restrict__ bias, const float* __restrict__ add,
            float* __restrict__ C, int M, int K, int N)
{
    const int lane = threadIdx.x;
    const int m0 = blockIdx.x << 4;
    const int n0 = blockIdx.y << 4;
    const int row  = lane & 15;
    const int half = lane >> 4;
    const int gr = m0 + row;          // A row handled by this lane
    const int gn = n0 + row;          // B column handled by this lane
    const bool fullA = (m0 + 16 <= M);    // wave-uniform
    v8f acc = {};
    for (int kk = 0; kk < K; kk += 32) {
        v16h af, bf;
        if (fullA && (kk + 32 <= K)) {        // wave-uniform fast path
            // A frag: contiguous k = kk+16h .. +15 (matches B)
            const float* ap = A + (size_t)gr * K + kk + (half << 4);
            f4 a0 = *(const f4*)(ap);
            f4 a1 = *(const f4*)(ap + 4);
            f4 a2 = *(const f4*)(ap + 8);
            f4 a3 = *(const f4*)(ap + 12);
            #pragma unroll
            for (int t = 0; t < 4; ++t) {
                af[t]      = (h16)a0[t];
                af[4 + t]  = (h16)a1[t];
                af[8 + t]  = (h16)a2[t];
                af[12 + t] = (h16)a3[t];
            }
            // B frag: 16 contiguous halves of transposed weights
            bf = *(const v16h*)(Wt + (size_t)gn * K + kk + (half << 4));
        } else {                               // tail tiles / K=16 GEMM
            #pragma unroll
            for (int idx = 0; idx < 16; ++idx) {
                int ka = kk + (half << 4) + idx;
                float av = (gr < M && ka < K) ? A[(size_t)gr * K + ka] : 0.f;
                af[idx] = (h16)av;
                int kb = kk + (half << 4) + idx;
                bf[idx] = (kb < K) ? Wt[(size_t)gn * K + kb] : (h16)0.f;
            }
        }
        acc = __builtin_amdgcn_wmma_f32_16x16x32_f16(false, af, false, bf,
                                                     (short)0, acc, false, false);
    }
    // C layout: lane l -> col (l&15); VGPR r -> row r + 8*(l>>4)
    const int n = n0 + row;
    #pragma unroll
    for (int r = 0; r < 8; ++r) {
        int gm = m0 + r + (half << 3);
        if (gm < M) {
            float v = acc[r];
            if (bias) v += bias[n];
            if (add)  v += add[gm * N + n];
            if (ACT == 1) v = reluf(v);
            else if (ACT == 2) v = eluf(v);
            C[gm * N + n] = v;
        }
    }
}

// ---------------------------------------------------------------------------
// graph bookkeeping (batch is sorted -> contiguous per-graph node ranges)
// ---------------------------------------------------------------------------
__global__ void k_graph_bounds(const int* __restrict__ batch, int n, int* __restrict__ start){
    int i = blockIdx.x * blockDim.x + threadIdx.x;
    if (i >= n) return;
    int b = batch[i];
    if (i == 0) { for (int g = 0; g <= b; ++g) start[g] = 0; }
    else {
        int pb = batch[i - 1];
        for (int g = pb + 1; g <= b; ++g) start[g] = i;
    }
    if (i == n - 1) { for (int g = b + 1; g <= NG; ++g) start[g] = n; }
}

__global__ void k_hist(const int* __restrict__ dst, int ne, int* __restrict__ deg){
    int i = blockIdx.x * blockDim.x + threadIdx.x;
    if (i < ne) atomicAdd(&deg[dst[i]], 1);
}
__global__ void k_scan(const int* __restrict__ deg, int n, int* __restrict__ off){
    if (blockIdx.x == 0 && threadIdx.x == 0) {
        int a = 0;
        for (int i = 0; i < n; ++i) { off[i] = a; a += deg[i]; }
        off[n] = a;
    }
}
__global__ void k_fill(const int* __restrict__ dst, int ne,
                       int* __restrict__ cursor, int* __restrict__ elist){
    int i = blockIdx.x * blockDim.x + threadIdx.x;
    if (i < ne) { int pos = atomicAdd(&cursor[dst[i]], 1); elist[pos] = i; }
}

// ---------------------------------------------------------------------------
// graph-LN (scalar mean/var per graph over nodes*channels) + ELU, fused
// ---------------------------------------------------------------------------
__global__ void k_graph_ln_elu(const float* __restrict__ xin, const int* __restrict__ gstart,
                               const float* __restrict__ w, const float* __restrict__ b,
                               float* __restrict__ xout){
    __shared__ float r1[256], r2[256];
    int g = blockIdx.x, tid = threadIdx.x;
    int s = gstart[g], e = gstart[g + 1];
    int cnt = (e - s) * DN;
    float sum = 0.f, sq = 0.f;
    for (int i = tid; i < cnt; i += 256) {
        float v = xin[s * DN + i];
        sum += v; sq += v * v;
    }
    r1[tid] = sum; r2[tid] = sq; __syncthreads();
    for (int o = 128; o > 0; o >>= 1) {
        if (tid < o) { r1[tid] += r1[tid + o]; r2[tid] += r2[tid + o]; }
        __syncthreads();
    }
    float c    = (float)(cnt < 1 ? 1 : cnt);
    float mean = r1[0] / c;
    float var  = fmaxf(r2[0] / c - mean * mean, 0.f);
    float inv  = rsqrtf(var + 1e-5f);
    for (int i = tid; i < cnt; i += 256) {
        int ch = i & (DN - 1);
        float v = (xin[s * DN + i] - mean) * inv * w[ch] + b[ch];
        xout[s * DN + i] = eluf(v);
    }
}

// ---------------------------------------------------------------------------
// concat gathers
// ---------------------------------------------------------------------------
__global__ void k_edge_cat(const float* __restrict__ x, const float* __restrict__ ea,
                           const int* __restrict__ src, float* __restrict__ ecat, int ne){
    int i = blockIdx.x * blockDim.x + threadIdx.x;
    if (i >= ne * 2 * DN) return;
    int j = i >> 7, c = i & 127;
    ecat[i] = (c < DN) ? x[src[j] * DN + c] : ea[j * DN + (c - DN)];
}
__global__ void k_node_cat(const float* __restrict__ x, const float* __restrict__ nm,
                           float* __restrict__ xcat, int nn){
    int i = blockIdx.x * blockDim.x + threadIdx.x;
    if (i >= nn * 2 * DN) return;
    int n = i >> 7, c = i & 127;
    xcat[i] = (c < DN) ? x[n * DN + c] : nm[n * DN + (c - DN)];
}

// ---------------------------------------------------------------------------
// sparse replacement for einsum('ij,id->jd', valid_f, eh):
// m[j] = sum over in-edges i of node src[j], excluding reverse edges
// ---------------------------------------------------------------------------
__global__ void k_edge_msg(const float* __restrict__ eh, const int* __restrict__ src,
                           const int* __restrict__ dst, const int* __restrict__ off,
                           const int* __restrict__ elist, float* __restrict__ m){
    int j = blockIdx.x;       // edge
    int c = threadIdx.x;      // channel 0..63
    int s = src[j], d = dst[j];
    float acc = 0.f;
    int p1 = off[s + 1];
    for (int p = off[s]; p < p1; ++p) {
        int i = elist[p];
        if (src[i] != d) acc += eh[i * DN + c];
    }
    m[j * DN + c] = acc;
}

__global__ void k_scatter_dst(const float* __restrict__ eh, const int* __restrict__ dst,
                              float* __restrict__ out, int ne){
    int i = blockIdx.x * blockDim.x + threadIdx.x;
    if (i >= ne * DN) return;
    int j = i >> 6, c = i & (DN - 1);
    atomicAdd(&out[dst[j] * DN + c], eh[i]);
}
__global__ void k_scatter_src_dst(const float* __restrict__ x, const int* __restrict__ src,
                                  const int* __restrict__ dst, float* __restrict__ out, int ne){
    int i = blockIdx.x * blockDim.x + threadIdx.x;
    if (i >= ne * DN) return;
    int j = i >> 6, c = i & (DN - 1);
    atomicAdd(&out[dst[j] * DN + c], x[src[j] * DN + c]);
}

// ---------------------------------------------------------------------------
// SAG pooling: score, per-graph softmax + weighted global-add-pool
// ---------------------------------------------------------------------------
__global__ void k_sag_score(const float* __restrict__ agg, const float* __restrict__ x,
                            const float* __restrict__ relW, const float* __restrict__ relb,
                            const float* __restrict__ rootW, float* __restrict__ score, int nn){
    int n = blockIdx.x * blockDim.x + threadIdx.x;
    if (n >= nn) return;
    float s = relb[0];
    for (int c = 0; c < DN; ++c)
        s += agg[n * DN + c] * relW[c] + x[n * DN + c] * rootW[c];
    score[n] = s;
}

__global__ void k_pool(const float* __restrict__ x, const float* __restrict__ score,
                       const int* __restrict__ gstart, float* __restrict__ g){
    __shared__ float red[128];
    __shared__ float smax, ssum;
    int gr = blockIdx.x, tid = threadIdx.x;
    int s = gstart[gr], e = gstart[gr + 1];
    float mx = -3.402823466e38f;
    for (int n = s + tid; n < e; n += 128) mx = fmaxf(mx, score[n]);
    red[tid] = mx; __syncthreads();
    for (int o = 64; o > 0; o >>= 1) { if (tid < o) red[tid] = fmaxf(red[tid], red[tid + o]); __syncthreads(); }
    if (tid == 0) smax = red[0];
    __syncthreads();
    float sum = 0.f;
    for (int n = s + tid; n < e; n += 128) sum += __expf(score[n] - smax);
    red[tid] = sum; __syncthreads();
    for (int o = 64; o > 0; o >>= 1) { if (tid < o) red[tid] += red[tid + o]; __syncthreads(); }
    if (tid == 0) ssum = fmaxf(red[0], 1e-16f);
    __syncthreads();
    if (tid < DN) {
        float acc = 0.f;
        for (int n = s; n < e; ++n)
            acc += x[n * DN + tid] * (__expf(score[n] - smax) / ssum);
        g[gr * DN + tid] = acc;
    }
}

__global__ void k_l2norm_store(const float* __restrict__ in, float* __restrict__ out, int ostride){
    __shared__ float red[DHID];
    int g = blockIdx.x, c = threadIdx.x;
    float v = in[g * DHID + c];
    red[c] = v * v; __syncthreads();
    for (int o = DHID / 2; o > 0; o >>= 1) { if (c < o) red[c] += red[c + o]; __syncthreads(); }
    float nrm = fmaxf(sqrtf(red[0]), 1e-12f);
    out[g * ostride + c] = v / nrm;
}

// ---------------------------------------------------------------------------
// co-attention pieces
// ---------------------------------------------------------------------------
__global__ void k_ln_rows(const float* __restrict__ in, const float* __restrict__ w,
                          const float* __restrict__ b, float* __restrict__ out){
    __shared__ float r1[DHID], r2[DHID];
    int r = blockIdx.x, c = threadIdx.x;
    float v = in[r * DHID + c];
    r1[c] = v; r2[c] = v * v; __syncthreads();
    for (int o = DHID / 2; o > 0; o >>= 1) { if (c < o) { r1[c] += r1[c + o]; r2[c] += r2[c + o]; } __syncthreads(); }
    float mean = r1[0] / (float)DHID;
    float var  = fmaxf(r2[0] / (float)DHID - mean * mean, 0.f);
    out[r * DHID + c] = (v - mean) * rsqrtf(var + 1e-5f) * w[c] + b[c];
}

__global__ void k_coattn(const float* __restrict__ q, const float* __restrict__ kv,
                         float* __restrict__ out){
    __shared__ float red[DHEAD];
    __shared__ float attn[NB][NB];
    int g = blockIdx.x, h = blockIdx.y, d = threadIdx.x;
    float qv[NB], kk[NB], vv[NB];
    for (int n = 0; n < NB; ++n)
        qv[n] = q[(g * NB + n) * (HEADS * DHEAD) + h * DHEAD + d];
    for (int m = 0; m < NB; ++m) {
        kk[m] = kv[(g * NB + m) * (2 * HEADS * DHEAD) + h * DHEAD + d];
        vv[m] = kv[(g * NB + m) * (2 * HEADS * DHEAD) + HEADS * DHEAD + h * DHEAD + d];
    }
    const float scale = 0.08838834764831845f;  // 128^-0.5
    for (int n = 0; n < NB; ++n)
        for (int m = 0; m < NB; ++m) {
            red[d] = qv[n] * kk[m];
            __syncthreads();
            for (int o = DHEAD / 2; o > 0; o >>= 1) { if (d < o) red[d] += red[d + o]; __syncthreads(); }
            if (d == 0) attn[n][m] = red[0] * scale;
            __syncthreads();
        }
    if (d == 0) {
        for (int n = 0; n < NB; ++n) {
            float mx = attn[n][0];
            for (int m = 1; m < NB; ++m) mx = fmaxf(mx, attn[n][m]);
            float s = 0.f;
            for (int m = 0; m < NB; ++m) { attn[n][m] = __expf(attn[n][m] - mx); s += attn[n][m]; }
            for (int m = 0; m < NB; ++m) attn[n][m] /= s;
        }
    }
    __syncthreads();
    for (int n = 0; n < NB; ++n) {
        float o = 0.f;
        for (int m = 0; m < NB; ++m) o += attn[n][m] * vv[m];
        out[(g * NB + n) * (HEADS * DHEAD) + h * DHEAD + d] = o;
    }
}

__global__ void k_final(const float* __restrict__ ha, const float* __restrict__ ta,
                        const float* __restrict__ m1W, const float* __restrict__ m1b,
                        const float* __restrict__ m2W, const float* __restrict__ m2b,
                        float* __restrict__ out){
    __shared__ float red[DHID];
    __shared__ float sim[NB * NB];
    int g = blockIdx.x, d = threadIdx.x;
    for (int n = 0; n < NB; ++n)
        for (int m = 0; m < NB; ++m) {
            red[d] = ha[(g * NB + n) * DHID + d] * ta[(g * NB + m) * DHID + d];
            __syncthreads();
            for (int o = DHID / 2; o > 0; o >>= 1) { if (d < o) red[d] += red[d + o]; __syncthreads(); }
            if (d == 0) sim[n * NB + m] = red[0];
            __syncthreads();
        }
    if (d == 0) {
        float h0 = m1b[0], h1 = m1b[1];
        for (int k2 = 0; k2 < NB * NB; ++k2) { h0 += sim[k2] * m1W[k2 * 2 + 0]; h1 += sim[k2] * m1W[k2 * 2 + 1]; }
        h0 = reluf(h0); h1 = reluf(h1);
        out[g] = h0 * m2W[0] + h1 * m2W[1] + m2b[0];
    }
}

// ---------------------------------------------------------------------------
// host orchestration
// ---------------------------------------------------------------------------
extern "C" void kernel_launch(void* const* d_in, const int* in_sizes, int n_in,
                              void* d_out, int out_size, void* d_ws, size_t ws_size,
                              hipStream_t stream)
{
    (void)in_sizes; (void)n_in; (void)out_size; (void)ws_size;
    auto F = [&](int i) -> const float* { return (const float*)d_in[i]; };
    auto I = [&](int i) -> const int*   { return (const int*)d_in[i]; };
    const int PB = 23;
    auto BP = [&](int blk, int k) -> const float* { return F(PB + 24 * blk + k); };

    char* base = (char*)d_ws;
    size_t off_b = 0;
    auto bump = [&](size_t bytes) -> void* {
        void* p = base + off_b;
        off_b = (off_b + bytes + 255) & ~(size_t)255;
        return p;
    };
    auto fbuf = [&](size_t n) -> float* { return (float*)bump(n * sizeof(float)); };
    auto ibuf = [&](size_t n) -> int*   { return (int*)bump(n * sizeof(int)); };
    // convert [KxN] f32 weight to transposed [NxK] f16
    auto conv = [&](const float* w, int K, int N) -> const h16* {
        h16* d = (h16*)bump((size_t)K * N * sizeof(h16));
        int n = K * N;
        k_f32h_t<<<(n + 255) / 256, 256, 0, stream>>>(w, d, K, N);
        return d;
    };

    // f16 transposed weights
    const h16* w_init_node = conv(F(8), 64, 64);
    const h16* w_init_edge = conv(F(10), 16, 64);
    const h16* w_att_q   = conv(F(16), 128, 256);
    const h16* w_att_kv  = conv(F(17), 128, 512);
    const h16* w_att_out = conv(F(18), 256, 128);
    const h16 *w_lin[NB], *w_lin2[NB], *w_c1e[NB], *w_c1m[NB], *w_c1n[NB],
              *w_c2e[NB], *w_c2m[NB], *w_c2n[NB], *w_trans[NB];
    for (int i = 0; i < NB; ++i) {
        w_lin[i]   = conv(BP(i, 0), 64, 64);
        w_lin2[i]  = conv(BP(i, 2), 64, 64);
        w_c1e[i]   = conv(BP(i, 4), 128, 64);
        w_c1m[i]   = conv(BP(i, 6), 64, 64);
        w_c1n[i]   = conv(BP(i, 8), 128, 64);
        w_c2e[i]   = conv(BP(i, 10), 128, 64);
        w_c2m[i]   = conv(BP(i, 12), 64, 64);
        w_c2n[i]   = conv(BP(i, 14), 128, 64);
        w_trans[i] = conv(BP(i, 23), 64, 128);
    }

    // scratch
    int*   gstart = ibuf(NG + 1);
    int*   deg    = ibuf(NN);
    int*   offs   = ibuf(NN + 1);
    int*   cursor = ibuf(NN);
    int*   elist  = ibuf(NE);
    float* x      = fbuf(NN * DN);
    float* xtmp   = fbuf(NN * DN);
    float* eaA    = fbuf(NE * DN);
    float* eaB    = fbuf(NE * DN);
    float* ecat   = fbuf(NE * 2 * DN);
    float* eh0    = fbuf(NE * DN);
    float* eh     = fbuf(NE * DN);
    float* mbuf   = fbuf(NE * DN);
    float* nmsg   = fbuf(NN * DN);
    float* xcat   = fbuf(NN * 2 * DN);
    float* agg    = fbuf(NN * DN);
    float* score  = fbuf(NN);
    float* gpool  = fbuf(NG * DN);
    float* tpool  = fbuf(NG * DHID);
    float* rh     = fbuf(NG * NB * DHID);
    float* rt     = fbuf(NG * NB * DHID);
    float* rhn    = fbuf(NG * NB * DHID);
    float* rtn    = fbuf(NG * NB * DHID);
    float* qb     = fbuf(NG * NB * HEADS * DHEAD);
    float* kvb    = fbuf(NG * NB * 2 * HEADS * DHEAD);
    float* ob     = fbuf(NG * NB * HEADS * DHEAD);
    float* ha     = fbuf(NG * NB * DHID);
    float* ta     = fbuf(NG * NB * DHID);

    auto gemm = [&](const float* A, const h16* W, const float* bias, const float* add,
                    float* C, int M, int K, int N, int act) {
        dim3 grid((M + 15) / 16, N / 16);
        switch (act) {
        case 1:  k_wmma_gemm<1><<<grid, 32, 0, stream>>>(A, W, bias, add, C, M, K, N); break;
        case 2:  k_wmma_gemm<2><<<grid, 32, 0, stream>>>(A, W, bias, add, C, M, K, N); break;
        default: k_wmma_gemm<0><<<grid, 32, 0, stream>>>(A, W, bias, add, C, M, K, N); break;
        }
    };

    auto side = [&](const float* x_in, const int* ei, const float* ea_in, const int* batch,
                    float* r_out) {
        const int* src = ei;
        const int* dst = ei + NE;
        k_graph_bounds<<<(NN + 255) / 256, 256, 0, stream>>>(batch, NN, gstart);
        hipMemsetAsync(deg, 0, NN * sizeof(int), stream);
        k_hist<<<(NE + 255) / 256, 256, 0, stream>>>(dst, NE, deg);
        k_scan<<<1, 1, 0, stream>>>(deg, NN, offs);
        hipMemcpyAsync(cursor, offs, NN * sizeof(int), hipMemcpyDeviceToDevice, stream);
        k_fill<<<(NE + 255) / 256, 256, 0, stream>>>(dst, NE, cursor, elist);

        // init node / edge
        gemm(x_in, w_init_node, F(9), nullptr, xtmp, NN, DN, DN, 0);
        k_graph_ln_elu<<<NG, 256, 0, stream>>>(xtmp, gstart, F(12), F(13), x);
        gemm(ea_in, w_init_edge, F(11), nullptr, eaA, NE, DEI, DN, 2);
        float* ea   = eaA;
        float* ealt = eaB;

        auto dmpnn = [&](const h16* we, const float* be, const h16* wm, const float* bm,
                         const h16* wn, const float* bn) {
            k_edge_cat<<<(NE * 2 * DN + 255) / 256, 256, 0, stream>>>(x, ea, src, ecat, NE);
            gemm(ecat, we, be, nullptr, eh0, NE, 2 * DN, DN, 1);
            hipMemcpyAsync(eh, eh0, NE * DN * sizeof(float), hipMemcpyDeviceToDevice, stream);
            for (int t = 0; t < TST; ++t) {
                k_edge_msg<<<NE, DN, 0, stream>>>(eh, src, dst, offs, elist, mbuf);
                gemm(mbuf, wm, bm, eh0, eh, NE, DN, DN, 1);   // fused eh0 add + relu
            }
            hipMemsetAsync(nmsg, 0, NN * DN * sizeof(float), stream);
            k_scatter_dst<<<(NE * DN + 255) / 256, 256, 0, stream>>>(eh, dst, nmsg, NE);
            k_node_cat<<<(NN * 2 * DN + 255) / 256, 256, 0, stream>>>(x, nmsg, xcat, NN);
            gemm(xcat, wn, bn, nullptr, xtmp, NN, 2 * DN, DN, 1);
        };

        for (int i = 0; i < NB; ++i) {
            gemm(ea, w_lin[i], BP(i, 1), nullptr, ealt, NE, DN, DN, 2);
            { float* t_ = ea; ea = ealt; ealt = t_; }
            dmpnn(w_c1e[i], BP(i, 5), w_c1m[i], BP(i, 7), w_c1n[i], BP(i, 9));
            k_graph_ln_elu<<<NG, 256, 0, stream>>>(xtmp, gstart, BP(i, 16), BP(i, 17), x);
            gemm(ea, w_lin2[i], BP(i, 3), nullptr, ealt, NE, DN, DN, 2);
            { float* t_ = ea; ea = ealt; ealt = t_; }
            dmpnn(w_c2e[i], BP(i, 11), w_c2m[i], BP(i, 13), w_c2n[i], BP(i, 15));
            k_graph_ln_elu<<<NG, 256, 0, stream>>>(xtmp, gstart, BP(i, 18), BP(i, 19), x);
            // SAG pooling + projection + l2norm
            hipMemsetAsync(agg, 0, NN * DN * sizeof(float), stream);
            k_scatter_src_dst<<<(NE * DN + 255) / 256, 256, 0, stream>>>(x, src, dst, agg, NE);
            k_sag_score<<<(NN + 255) / 256, 256, 0, stream>>>(agg, x, BP(i, 20), BP(i, 21),
                                                              BP(i, 22), score, NN);
            k_pool<<<NG, 128, 0, stream>>>(x, score, gstart, gpool);
            gemm(gpool, w_trans[i], nullptr, nullptr, tpool, NG, DN, DHID, 0);
            k_l2norm_store<<<NG, DHID, 0, stream>>>(tpool, r_out + i * DHID, NB * DHID);
        }
    };

    side(F(0), I(1), F(2), I(3), rh);
    side(F(4), I(5), F(6), I(7), rt);

    // co-attention both directions
    k_ln_rows<<<NG * NB, DHID, 0, stream>>>(rh, F(14), F(15), rhn);
    k_ln_rows<<<NG * NB, DHID, 0, stream>>>(rt, F(14), F(15), rtn);

    gemm(rhn, w_att_q,  nullptr, nullptr, qb,  NG * NB, DHID, HEADS * DHEAD, 0);
    gemm(rtn, w_att_kv, nullptr, nullptr, kvb, NG * NB, DHID, 2 * HEADS * DHEAD, 0);
    { dim3 g(NG, HEADS); k_coattn<<<g, DHEAD, 0, stream>>>(qb, kvb, ob); }
    gemm(ob, w_att_out, nullptr, nullptr, ha, NG * NB, HEADS * DHEAD, DHID, 0);

    gemm(rtn, w_att_q,  nullptr, nullptr, qb,  NG * NB, DHID, HEADS * DHEAD, 0);
    gemm(rhn, w_att_kv, nullptr, nullptr, kvb, NG * NB, DHID, 2 * HEADS * DHEAD, 0);
    { dim3 g(NG, HEADS); k_coattn<<<g, DHEAD, 0, stream>>>(qb, kvb, ob); }
    gemm(ob, w_att_out, nullptr, nullptr, ta, NG * NB, HEADS * DHEAD, DHID, 0);

    k_final<<<NG, DHID, 0, stream>>>(ha, ta, F(19), F(20), F(21), F(22), (float*)d_out);
}
